// CosAttentionNet_39591008535145
// MI455X (gfx1250) — matrix operand order repaired and checked by reading, not compiled, not run on hardware
//
#include <hip/hip_runtime.h>
#include <hip/hip_bf16.h>

// ---------------------------------------------------------------------------
// CDNA5 (gfx1250) implementation of CosAttentionNet forward.
// All dense GEMMs use v_wmma_f32_16x16x32_bf16 (f32 accumulate).
// GRU recurrences run as persistent workgroups with w_hh / h / gate
// pre-activations resident in LDS (240 KB of the WGP's 320 KB).
// All WMMA B operands are n-major ([N,K] row-major) so every fragment load
// vectorizes to global_load_b128 / ds_load_b128 pairs; the one k-major
// operand (ctx_outs in the second attention bmm) is transposed once in LDS.
// ---------------------------------------------------------------------------

typedef __attribute__((ext_vector_type(16))) __bf16 v16bf;
typedef __attribute__((ext_vector_type(8)))  float  v8f;

union Frag16 { v16bf v; unsigned short u[16]; };

__device__ __forceinline__ unsigned short f2bfbits(float x) {
    union { float f; unsigned u; } v; v.f = x;
    unsigned r = (v.u + 0x7FFFu + ((v.u >> 16) & 1u)) >> 16;
    return (unsigned short)r;
}

// ---------------- constants --------------------------------------------------
#define BSZ   64
#define L1S   512
#define L2S   50
#define L2P   64
#define NOPT  10
#define DIN   300
#define DPAD  320
#define HS    128
#define GS    384      // 3H
#define H2S   256      // 2H
#define H4S   512      // 4H
#define BN    (BSZ*NOPT)          // 640
#define ROWS_CTX (BSZ*L1S)        // 32768
#define ROWS_OPT (BN*L2S)         // 32000

// ---------------- fp32 -> bf16 convert with K padding ------------------------
__global__ void cvt_pad_bf16_kernel(const float* __restrict__ src,
                                    unsigned short* __restrict__ dst,
                                    int rows, int ksrc, int kdst) {
    long t = (long)blockIdx.x * blockDim.x + threadIdx.x;
    long total = (long)rows * kdst;
    if (t >= total) return;
    int k = (int)(t % kdst);
    long r = t / kdst;
    float v = (k < ksrc) ? src[r * ksrc + k] : 0.0f;
    dst[t] = f2bfbits(v);
}

// ---------------- row-wise L2 normalize (width 256) -> bf16 ------------------
__global__ void rownorm_bf16_kernel(const float* __restrict__ src,
                                    unsigned short* __restrict__ dst) {
    __shared__ float red[256];
    long row = blockIdx.x;
    int tid = threadIdx.x;
    float x = src[row * H2S + tid];
    red[tid] = x * x;
    __syncthreads();
    for (int s = 128; s > 0; s >>= 1) {
        if (tid < s) red[tid] += red[tid + s];
        __syncthreads();
    }
    float inv = 1.0f / fmaxf(sqrtf(red[0]), 1e-8f);
    dst[row * H2S + tid] = f2bfbits(x * inv);
}

// ---------------- LDS-tiled transpose: f32 [b][512][256] -> bf16 [b][256][512]
__global__ __launch_bounds__(256)
void transpose_bf16_kernel(const float* __restrict__ src,
                           unsigned short* __restrict__ dst) {
    __shared__ unsigned short tile[16][17];
    int tx = threadIdx.x & 15;
    int ty = threadIdx.x >> 4;
    int k0 = blockIdx.x * 16;          // over L1 (512)
    int n0 = blockIdx.y * 16;          // over 2H (256)
    long b = blockIdx.z;
    tile[ty][tx] = f2bfbits(src[(b * L1S + k0 + ty) * H2S + n0 + tx]);
    __syncthreads();
    dst[(b * H2S + n0 + ty) * L1S + k0 + tx] = tile[tx][ty];
}

// ---------------- softmax over 512 (attention rows) -> bf16 ------------------
__global__ void softmax512_bf16_kernel(const float* __restrict__ att,
                                       unsigned short* __restrict__ att_bf) {
    __shared__ float red[256];
    long row = blockIdx.x;
    int tid = threadIdx.x;
    float v0 = att[row * 512 + tid];
    float v1 = att[row * 512 + 256 + tid];
    red[tid] = fmaxf(v0, v1);
    __syncthreads();
    for (int s = 128; s > 0; s >>= 1) {
        if (tid < s) red[tid] = fmaxf(red[tid], red[tid + s]);
        __syncthreads();
    }
    float m = red[0];
    __syncthreads();
    float e0 = __expf(v0 - m), e1 = __expf(v1 - m);
    red[tid] = e0 + e1;
    __syncthreads();
    for (int s = 128; s > 0; s >>= 1) {
        if (tid < s) red[tid] += red[tid + s];
        __syncthreads();
    }
    float inv = 1.0f / red[0];
    att_bf[row * 512 + tid]       = f2bfbits(e0 * inv);
    att_bf[row * 512 + 256 + tid] = f2bfbits(e1 * inv);
}

// ---------------- concat [att_opt | opt_outs] -> bf16 [32000,512] ------------
__global__ void build_cat_kernel(const float* __restrict__ att_opt,   // [640,64,256]
                                 const float* __restrict__ opt_outs,  // [640,64,256]
                                 unsigned short* __restrict__ cat_bf) // [640*50,512]
{
    long t = (long)blockIdx.x * blockDim.x + threadIdx.x;
    if (t >= (long)ROWS_OPT * H4S) return;
    int col = (int)(t % H4S);
    long row = t / H4S;           // i*50 + q
    long i = row / L2S;
    long q = row - i * L2S;
    float v;
    if (col < H2S) v = att_opt[(i * L2P + q) * H2S + col];
    else           v = opt_outs[(i * L2P + q) * H2S + (col - H2S)];
    cat_bf[t] = f2bfbits(v);
}

// ---------------- generic batched bf16 WMMA GEMM -----------------------------
// C[M,N] = A[M,K] * B^T + bias
// A row-major [M,lda]; B n-major [N,ldb] row-major (B[k][n] = Bm[n*ldb+k]).
__global__ __launch_bounds__(128)
void wmma_gemm_bf16_kernel(const unsigned short* __restrict__ A,
                           const unsigned short* __restrict__ Bm,
                           const float* __restrict__ bias,
                           float* __restrict__ C,
                           int Mtiles, int K, int lda, int ldb, int ldc,
                           long sA, long sB, long sC, int bDivB) {
    int tid  = threadIdx.x;
    int wave = tid >> 5;
    int mtile = blockIdx.x * 4 + wave;
    if (mtile >= Mtiles) return;
    int ntile = blockIdx.y;
    long z = blockIdx.z;

    int lane = tid & 31;
    int half = lane >> 4;
    int r    = lane & 15;
    int m0 = mtile * 16;
    int n0 = ntile * 16;

    const unsigned short* Ap = A  + z * sA + (long)(m0 + r) * lda;
    const unsigned short* Bp = Bm + (z / bDivB) * sB + (long)(n0 + r) * ldb + half * 16;
    float* Cb = C + z * sC;

    v8f acc = {0.f,0.f,0.f,0.f,0.f,0.f,0.f,0.f};
    for (int k0 = 0; k0 < K; k0 += 32) {
        // A fragment: 16x32 bf16 (one contiguous 32B chunk per lane)
        Frag16 fa;
        {
            const unsigned short* rp = Ap + k0 + half * 16;
            #pragma unroll
            for (int i = 0; i < 8; ++i) fa.u[i]     = rp[i];
            #pragma unroll
            for (int i = 0; i < 8; ++i) fa.u[8 + i] = rp[8 + i];
        }
        // B fragment: 32x16 bf16 (one contiguous 32B chunk per lane)
        Frag16 fb;
        {
            const unsigned short* rp = Bp + k0;
            #pragma unroll
            for (int i = 0; i < 16; ++i) fb.u[i] = rp[i];
        }
        acc = __builtin_amdgcn_wmma_f32_16x16x32_bf16(false, fa.v, false, fb.v,
                                                      (short)0, acc, false, false);
    }
    #pragma unroll
    for (int i = 0; i < 8; ++i) {
        int row = m0 + half * 8 + i;
        int col = n0 + r;
        float v = acc[i] + (bias ? bias[col] : 0.0f);
        Cb[(long)row * ldc + col] = v;
    }
}

// ---------------- persistent GRU recurrence ---------------------------------
// One 256-thread workgroup processes 64 batch rows for one direction.
// LDS: w_hh bf16 [384][128] (96KB) + gh f32 [64][384] (96KB)
//    + h f32 [64][128] (32KB)      + h bf16 [64][128] (16KB)  = 240KB
#define GRU_SMEM_BYTES (GS*HS*2 + 64*GS*4 + 64*HS*4 + 64*HS*2)

__global__ __launch_bounds__(256)
void gru_recurrence_kernel(const float* __restrict__ gx,          // [rows, L, 384]
                           const unsigned short* __restrict__ whh,// [384,128] bf16
                           const float* __restrict__ bhh,         // [384]
                           float* __restrict__ outs,              // [rows, Lout, ldOut] or null
                           float* __restrict__ hout,              // [rows, 256] or null
                           int L, int Lout, int ldOut, int colOff, int reverse) {
    extern __shared__ char smem[];
    unsigned short* wlds = (unsigned short*)smem;                        // [384][128]
    float* gh  = (float*)(smem + GS * HS * 2);                           // [64][384]
    float* hst = (float*)(smem + GS * HS * 2 + 64 * GS * 4);             // [64][128]
    unsigned short* hbf = (unsigned short*)(smem + GS * HS * 2 + 64 * GS * 4
                                                 + 64 * HS * 4);         // [64][128]

    int tid = threadIdx.x;
    int rowBase = blockIdx.x * 64;

    for (int i = tid; i < GS * HS; i += 256) wlds[i] = whh[i];
    for (int i = tid; i < 64 * HS; i += 256) { hst[i] = 0.0f; hbf[i] = 0; }
    __syncthreads();

    int wave = tid >> 5;
    int lane = tid & 31;
    int half = lane >> 4;
    int r    = lane & 15;

    for (int t = 0; t < L; ++t) {
        int t_eff = reverse ? (L - 1 - t) : t;

        // -------- phase A: gh = h @ w_hh^T (WMMA out of LDS) --------
        for (int tile = wave; tile < 96; tile += 8) {   // 4 Mtiles x 24 Ntiles
            int mt = tile & 3;
            int nt = tile >> 2;
            int m0 = mt * 16, n0 = nt * 16;
            v8f acc = {0.f,0.f,0.f,0.f,0.f,0.f,0.f,0.f};
            #pragma unroll
            for (int kt = 0; kt < 4; ++kt) {
                int k0 = kt * 32;
                Frag16 fa;                               // h (bf16 LDS mirror)
                {
                    const unsigned short* rp = hbf + (m0 + r) * HS + k0 + half * 16;
                    #pragma unroll
                    for (int i = 0; i < 8; ++i) fa.u[i]     = rp[i];
                    #pragma unroll
                    for (int i = 0; i < 8; ++i) fa.u[8 + i] = rp[8 + i];
                }
                Frag16 fb;                               // w_hh (bf16 LDS, [N,K])
                {
                    const unsigned short* rp = wlds + (n0 + r) * HS + k0 + half * 16;
                    #pragma unroll
                    for (int i = 0; i < 16; ++i) fb.u[i] = rp[i];
                }
                acc = __builtin_amdgcn_wmma_f32_16x16x32_bf16(false, fa.v, false, fb.v,
                                                              (short)0, acc, false, false);
            }
            #pragma unroll
            for (int i = 0; i < 8; ++i)
                gh[(m0 + half * 8 + i) * GS + n0 + r] = acc[i];
        }
        __syncthreads();

        // -------- phase B: gates + state update --------
        for (int e = tid; e < 64 * HS; e += 256) {
            int rr_ = e >> 7;       // local row
            int j   = e & 127;
            long gxb = ((long)(rowBase + rr_) * L + t_eff) * GS + j;
            float xr = gx[gxb], xz = gx[gxb + HS], xn = gx[gxb + 2 * HS];
            float hr = gh[rr_ * GS + j]          + bhh[j];
            float hz = gh[rr_ * GS + HS + j]     + bhh[HS + j];
            float hn = gh[rr_ * GS + 2 * HS + j] + bhh[2 * HS + j];
            float rg = 1.0f / (1.0f + __expf(-(xr + hr)));
            float zg = 1.0f / (1.0f + __expf(-(xz + hz)));
            float ng = tanhf(xn + rg * hn);
            float hp = hst[rr_ * HS + j];
            float hv = (1.0f - zg) * ng + zg * hp;
            hst[rr_ * HS + j] = hv;
            hbf[rr_ * HS + j] = f2bfbits(hv);
            if (outs)
                outs[((long)(rowBase + rr_) * Lout + t_eff) * ldOut + colOff + j] = hv;
        }
        __syncthreads();
    }

    if (hout) {
        for (int e = tid; e < 64 * HS; e += 256) {
            int rr_ = e >> 7, j = e & 127;
            hout[(long)(rowBase + rr_) * H2S + colOff + j] = hst[rr_ * HS + j];
        }
    }
}

// ---------------- final cosine + softmax over options ------------------------
__global__ void final_softmax_kernel(const float* __restrict__ ctx_h,  // [64,256]
                                     const float* __restrict__ ao_h,   // [640,256]
                                     float* __restrict__ out)          // [64,10]
{
    int b = threadIdx.x;
    if (b >= BSZ) return;
    float nc = 0.0f;
    for (int j = 0; j < H2S; ++j) { float v = ctx_h[b * H2S + j]; nc += v * v; }
    nc = fmaxf(sqrtf(nc), 1e-8f);
    float logits[NOPT];
    float m = -1e30f;
    for (int n = 0; n < NOPT; ++n) {
        const float* a = ao_h + ((long)b * NOPT + n) * H2S;
        float dot = 0.0f, na = 0.0f;
        for (int j = 0; j < H2S; ++j) {
            float cv = ctx_h[b * H2S + j], av = a[j];
            dot += cv * av; na += av * av;
        }
        float l = dot / (nc * fmaxf(sqrtf(na), 1e-8f));
        logits[n] = l;
        m = fmaxf(m, l);
    }
    float s = 0.0f;
    for (int n = 0; n < NOPT; ++n) { logits[n] = __expf(logits[n] - m); s += logits[n]; }
    float inv = 1.0f / s;
    for (int n = 0; n < NOPT; ++n) out[b * NOPT + n] = logits[n] * inv;
}

// ---------------------------------------------------------------------------
extern "C" void kernel_launch(void* const* d_in, const int* in_sizes, int n_in,
                              void* d_out, int out_size, void* d_ws, size_t ws_size,
                              hipStream_t stream) {
    (void)in_sizes; (void)n_in; (void)out_size; (void)ws_size;

    const float* context = (const float*)d_in[0];
    const float* options = (const float*)d_in[2];
    const float* rw_ih_f = (const float*)d_in[4];
    const float* rw_hh_f = (const float*)d_in[5];
    const float* rb_ih_f = (const float*)d_in[6];
    const float* rb_hh_f = (const float*)d_in[7];
    const float* rw_ih_b = (const float*)d_in[8];
    const float* rw_hh_b = (const float*)d_in[9];
    const float* rb_ih_b = (const float*)d_in[10];
    const float* rb_hh_b = (const float*)d_in[11];
    const float* aw_ih_f = (const float*)d_in[12];
    const float* aw_hh_f = (const float*)d_in[13];
    const float* ab_ih_f = (const float*)d_in[14];
    const float* ab_hh_f = (const float*)d_in[15];
    const float* aw_ih_b = (const float*)d_in[16];
    const float* aw_hh_b = (const float*)d_in[17];
    const float* ab_ih_b = (const float*)d_in[18];
    const float* ab_hh_b = (const float*)d_in[19];

    char* ws = (char*)d_ws;
    size_t off = 0;
    auto suballoc = [&](size_t bytes) -> char* {
        char* p = ws + off;
        off += (bytes + 255) & ~(size_t)255;
        return p;
    };

    // bf16 weight buffers
    unsigned short* rwihf_bf = (unsigned short*)suballoc((size_t)GS * DPAD * 2);
    unsigned short* rwihb_bf = (unsigned short*)suballoc((size_t)GS * DPAD * 2);
    unsigned short* rwhhf_bf = (unsigned short*)suballoc((size_t)GS * HS * 2);
    unsigned short* rwhhb_bf = (unsigned short*)suballoc((size_t)GS * HS * 2);
    unsigned short* awihf_bf = (unsigned short*)suballoc((size_t)GS * H4S * 2);
    unsigned short* awihb_bf = (unsigned short*)suballoc((size_t)GS * H4S * 2);
    unsigned short* awhhf_bf = (unsigned short*)suballoc((size_t)GS * HS * 2);
    unsigned short* awhhb_bf = (unsigned short*)suballoc((size_t)GS * HS * 2);
    // activations
    unsigned short* ctx_bf = (unsigned short*)suballoc((size_t)ROWS_CTX * DPAD * 2);
    unsigned short* opt_bf = (unsigned short*)suballoc((size_t)ROWS_OPT * DPAD * 2);
    // gx pools: reused across phases (ctx gx -> opt gx -> attn gx)
    float* gxP1 = (float*)suballoc((size_t)ROWS_CTX * GS * 4);
    float* gxP2 = (float*)suballoc((size_t)ROWS_CTX * GS * 4);
    float* ctx_outs = (float*)suballoc((size_t)ROWS_CTX * H2S * 4);
    float* ctx_h    = (float*)suballoc((size_t)BSZ * H2S * 4);
    unsigned short* ctx_norm_bf = (unsigned short*)suballoc((size_t)ROWS_CTX * H2S * 2);
    unsigned short* ctx_T_bf    = (unsigned short*)suballoc((size_t)ROWS_CTX * H2S * 2);
    float* opt_outs = (float*)suballoc((size_t)BN * L2P * H2S * 4);
    unsigned short* opt_norm_bf = (unsigned short*)suballoc((size_t)BN * L2P * H2S * 2);
    float* att = (float*)suballoc((size_t)BN * L2P * 512 * 4);
    unsigned short* att_bf = (unsigned short*)suballoc((size_t)BN * L2P * 512 * 2);
    float* att_opt = (float*)suballoc((size_t)BN * L2P * H2S * 4);
    unsigned short* cat_bf = (unsigned short*)suballoc((size_t)ROWS_OPT * H4S * 2);
    float* ao_h = (float*)suballoc((size_t)BN * H2S * 4);

    auto cvt = [&](const float* src, unsigned short* dst, int rows, int ksrc, int kdst) {
        long total = (long)rows * kdst;
        int blocks = (int)((total + 255) / 256);
        cvt_pad_bf16_kernel<<<blocks, 256, 0, stream>>>(src, dst, rows, ksrc, kdst);
    };

    // ---- 1. convert weights & activations to bf16 (K padded to mult. of 32)
    cvt(rw_ih_f, rwihf_bf, GS, DIN, DPAD);
    cvt(rw_ih_b, rwihb_bf, GS, DIN, DPAD);
    cvt(rw_hh_f, rwhhf_bf, GS, HS, HS);
    cvt(rw_hh_b, rwhhb_bf, GS, HS, HS);
    cvt(aw_ih_f, awihf_bf, GS, H4S, H4S);
    cvt(aw_ih_b, awihb_bf, GS, H4S, H4S);
    cvt(aw_hh_f, awhhf_bf, GS, HS, HS);
    cvt(aw_hh_b, awhhb_bf, GS, HS, HS);
    cvt(context, ctx_bf, ROWS_CTX, DIN, DPAD);
    cvt(options, opt_bf, ROWS_OPT, DIN, DPAD);

    // ---- 2. context input projections gx = x @ w_ih^T + b_ih (WMMA)
    wmma_gemm_bf16_kernel<<<dim3(ROWS_CTX/16/4, GS/16, 1), 128, 0, stream>>>(
        ctx_bf, rwihf_bf, rb_ih_f, gxP1, ROWS_CTX/16, DPAD, DPAD, DPAD, GS,
        0, 0, 0, 1);
    wmma_gemm_bf16_kernel<<<dim3(ROWS_CTX/16/4, GS/16, 1), 128, 0, stream>>>(
        ctx_bf, rwihb_bf, rb_ih_b, gxP2, ROWS_CTX/16, DPAD, DPAD, DPAD, GS,
        0, 0, 0, 1);

    // ---- 3. context BiGRU recurrence (persistent, LDS-resident)
    gru_recurrence_kernel<<<1, 256, GRU_SMEM_BYTES, stream>>>(
        gxP1, rwhhf_bf, rb_hh_f, ctx_outs, ctx_h, L1S, L1S, H2S, 0, 0);
    gru_recurrence_kernel<<<1, 256, GRU_SMEM_BYTES, stream>>>(
        gxP2, rwhhb_bf, rb_hh_b, ctx_outs, ctx_h, L1S, L1S, H2S, HS, 1);

    // ---- 4. option input projections (reuse gx pools)
    wmma_gemm_bf16_kernel<<<dim3(ROWS_OPT/16/4, GS/16, 1), 128, 0, stream>>>(
        opt_bf, rwihf_bf, rb_ih_f, gxP1, ROWS_OPT/16, DPAD, DPAD, DPAD, GS,
        0, 0, 0, 1);
    wmma_gemm_bf16_kernel<<<dim3(ROWS_OPT/16/4, GS/16, 1), 128, 0, stream>>>(
        opt_bf, rwihb_bf, rb_ih_b, gxP2, ROWS_OPT/16, DPAD, DPAD, DPAD, GS,
        0, 0, 0, 1);

    // ---- 5. option BiGRU recurrence (640 rows -> 10 persistent workgroups)
    gru_recurrence_kernel<<<BN/64, 256, GRU_SMEM_BYTES, stream>>>(
        gxP1, rwhhf_bf, rb_hh_f, opt_outs, nullptr, L2S, L2P, H2S, 0, 0);
    gru_recurrence_kernel<<<BN/64, 256, GRU_SMEM_BYTES, stream>>>(
        gxP2, rwhhb_bf, rb_hh_b, opt_outs, nullptr, L2S, L2P, H2S, HS, 1);

    // ---- 6. normalize (cosine attention) + transposed bf16 copy of ctx_outs
    rownorm_bf16_kernel<<<ROWS_CTX, 256, 0, stream>>>(ctx_outs, ctx_norm_bf);
    rownorm_bf16_kernel<<<BN * L2P, 256, 0, stream>>>(opt_outs, opt_norm_bf);
    transpose_bf16_kernel<<<dim3(L1S/16, H2S/16, BSZ), 256, 0, stream>>>(
        ctx_outs, ctx_T_bf);

    // ---- 7. att = normOpt @ normCtx^T  (batched WMMA; B shared per ctx batch)
    wmma_gemm_bf16_kernel<<<dim3(1, 512/16, BN), 128, 0, stream>>>(
        opt_norm_bf, ctx_norm_bf, nullptr, att, L2P/16, H2S, H2S, H2S, 512,
        (long)L2P * H2S, (long)L1S * H2S, (long)L2P * 512, NOPT);

    // ---- 8. softmax over context positions -> bf16
    softmax512_bf16_kernel<<<BN * L2P, 256, 0, stream>>>(att, att_bf);

    // ---- 9. att_opt = att @ ctx_outs  (uses transposed ctx_outs, n-major)
    wmma_gemm_bf16_kernel<<<dim3(1, H2S/16, BN), 128, 0, stream>>>(
        att_bf, ctx_T_bf, nullptr, att_opt, L2P/16, 512, 512, 512, H2S,
        (long)L2P * 512, (long)H2S * L1S, (long)L2P * H2S, NOPT);

    // ---- 10. cat = [att_opt | opt_outs] -> bf16 [32000, 512]
    {
        long total = (long)ROWS_OPT * H4S;
        build_cat_kernel<<<(int)((total + 255) / 256), 256, 0, stream>>>(
            att_opt, opt_outs, cat_bf);
    }

    // ---- 11. attention-GRU input projections (reuse gx pools)
    wmma_gemm_bf16_kernel<<<dim3(ROWS_OPT/16/4, GS/16, 1), 128, 0, stream>>>(
        cat_bf, awihf_bf, ab_ih_f, gxP1, ROWS_OPT/16, H4S, H4S, H4S, GS,
        0, 0, 0, 1);
    wmma_gemm_bf16_kernel<<<dim3(ROWS_OPT/16/4, GS/16, 1), 128, 0, stream>>>(
        cat_bf, awihb_bf, ab_ih_b, gxP2, ROWS_OPT/16, H4S, H4S, H4S, GS,
        0, 0, 0, 1);

    // ---- 12. attention BiGRU recurrence: only final hidden states needed
    gru_recurrence_kernel<<<BN/64, 256, GRU_SMEM_BYTES, stream>>>(
        gxP1, awhhf_bf, ab_hh_f, nullptr, ao_h, L2S, L2S, H2S, 0, 0);
    gru_recurrence_kernel<<<BN/64, 256, GRU_SMEM_BYTES, stream>>>(
        gxP2, awhhb_bf, ab_hh_b, nullptr, ao_h, L2S, L2S, H2S, HS, 1);

    // ---- 13. cosine(ctx_h, ao_h) + softmax over options
    final_softmax_kernel<<<1, 64, 0, stream>>>(ctx_h, ao_h, (float*)d_out);
}
